// DGT_47545287967266
// MI455X (gfx1250) — matrix operand order repaired and checked by reading, not compiled, not run on hardware
//
#include <hip/hip_runtime.h>
#include <hip/hip_bf16.h>
#include <math.h>

typedef __attribute__((ext_vector_type(16))) __bf16 v16bf;
typedef __attribute__((ext_vector_type(8)))  float  v8f;
typedef __attribute__((ext_vector_type(4)))  unsigned v4u;
typedef __attribute__((ext_vector_type(8)))  unsigned v8u;

#define P_TOT   262144            // 16*128*128 pixels
#define IMG     16384             // 128*128
#define MBYTES  (1024ull*1024ull)

__device__ __forceinline__ float bf2f(unsigned short h) {
    unsigned u = ((unsigned)h) << 16;
    return __builtin_bit_cast(float, u);
}
__device__ __forceinline__ unsigned short f2bf(float f) {
    unsigned u = __builtin_bit_cast(unsigned, f);
    unsigned r = u + 0x7FFFu + ((u >> 16) & 1u);
    return (unsigned short)(r >> 16);
}
__device__ __forceinline__ float sigmoidf(float x) { return 1.f / (1.f + expf(-x)); }

__device__ __forceinline__ v16bf cat_bf(v4u a, v4u b) {
    v8u c = __builtin_shufflevector(a, b, 0, 1, 2, 3, 4, 5, 6, 7);
    return __builtin_bit_cast(v16bf, c);
}

// ---------------- K0: fold BN params -------------------------------------
__global__ void params_k(const float* g1, const float* b1, const float* m1, const float* v1,
                         const float* g2, const float* b2, const float* m2, const float* v2,
                         const float* g3, const float* b3, const float* m3, const float* v3,
                         const float* cb, float* prm) {
    int c = threadIdx.x;              // 64 threads
    if (c >= 64) return;
    float s1 = g1[c] * rsqrtf(v1[c] + 1e-5f);
    float s2 = g2[c] * rsqrtf(v2[c] + 1e-5f);
    float s3 = g3[c] * rsqrtf(v3[c] + 1e-5f);
    prm[0*64 + c] = s1;  prm[1*64 + c] = b1[c] - m1[c] * s1;
    prm[2*64 + c] = s2;  prm[3*64 + c] = b2[c] - m2[c] * s2;
    prm[4*64 + c] = s3;  prm[5*64 + c] = b3[c] - m3[c] * s3;
    prm[6*64 + c] = 1.f; prm[7*64 + c] = cb[c];          // unit scale + combine bias
}

// ---------------- K0b: pack OIHW f32 weights -> [cout][tap*Cin] bf16 -----
__global__ void pack_w_k(const float* __restrict__ src, unsigned short* __restrict__ dst, int Cin) {
    int id = blockIdx.x * 256 + threadIdx.x;
    int K = Cin * 9;
    if (id >= 64 * K) return;
    int cout = id / K, r = id % K;
    int tap = r / Cin, cin = r % Cin;
    dst[id] = f2bf(src[(cout * Cin + cin) * 9 + tap]);
}

// ---------------- K1: x NCHW f32 -> NHWC bf16 (LDS tiled transpose) ------
__global__ __launch_bounds__(256) void nchw2nhwc_k(const float* __restrict__ src,
                                                   unsigned short* __restrict__ dst) {
    __shared__ unsigned short t[64][65];
    int blk = blockIdx.x;                 // 16 n * 256 r-tiles
    int n_ = blk >> 8;
    int r0 = (blk & 255) * 64;
    int rr = threadIdx.x & 63;
    int cg = threadIdx.x >> 6;            // 0..3
#pragma unroll
    for (int i = 0; i < 16; ++i) {
        int c = cg * 16 + i;
        t[c][rr] = f2bf(src[((size_t)(n_ * 64 + c) << 14) + r0 + rr]);
    }
    __syncthreads();
#pragma unroll
    for (int i = 0; i < 16; ++i) {
        int rloc = cg * 16 + i;
        dst[((size_t)(n_ << 14) + r0 + rloc) * 64 + rr] = t[rr][rloc];
    }
}

// ---------------- WMMA implicit-GEMM 3x3 conv ----------------------------
// in : NHWC bf16 [16][128][128][CIN], wpk: [64][9*CIN] bf16
// wave computes 16 pixels x 64 couts (4 accumulators sharing one A fragment)
template <int CIN>
__global__ __launch_bounds__(256) void conv3x3_wmma(const unsigned short* __restrict__ in,
                                                    const unsigned short* __restrict__ wpk,
                                                    const float* __restrict__ sc,
                                                    const float* __restrict__ bi,
                                                    unsigned short* __restrict__ obf,
                                                    float* __restrict__ of32, int doRelu) {
    const int K = CIN * 9;
    int lane = threadIdx.x & 31;
    int wave = threadIdx.x >> 5;
    int mtile = blockIdx.x * 8 + wave;
    int m0 = mtile * 16;

    // per-lane A pixel (tiles never cross rows: 128%16==0)
    int p   = m0 + (lane & 15);
    int n_  = p >> 14;
    int rem = p & (IMG - 1);
    int y = rem >> 7, x = rem & 127;
    int koff = (lane & 16) ? 8 : 0;     // A K-slot base (elements)
    int bko  = (lane & 16) ? 8 : 0;     // B K-slot base (elements)
    const unsigned short* wrow = wpk + (size_t)(lane & 15) * K;

    v8f acc[4];
#pragma unroll
    for (int nt = 0; nt < 4; ++nt)
#pragma unroll
        for (int i = 0; i < 8; ++i) acc[nt][i] = 0.f;

#pragma unroll
    for (int k0 = 0; k0 < K; k0 += 32) {
        const int tap  = k0 / CIN;
        const int cin0 = k0 % CIN;
        int iy = y + tap / 3 - 1, ix = x + tap % 3 - 1;
        int iyc = min(max(iy, 0), 127);
        int ixc = min(max(ix, 0), 127);
        unsigned msk = (iy == iyc && ix == ixc) ? 0xFFFFFFFFu : 0u;
        v4u mv = {msk, msk, msk, msk};
        const v4u* srcp = (const v4u*)(in + ((size_t)((n_ * 128 + iyc) * 128 + ixc) * CIN + cin0 + koff));
        v4u a0 = srcp[0] & mv;
        v4u a1 = srcp[2] & mv;                 // +16 bf16 = +2 v4u
        v16bf Av = cat_bf(a0, a1);
#pragma unroll
        for (int nt = 0; nt < 4; ++nt) {
            const v4u* wp = (const v4u*)(wrow + (size_t)nt * 16 * K + k0 + bko);
            v16bf Bv = cat_bf(wp[0], wp[2]);
            acc[nt] = __builtin_amdgcn_wmma_f32_16x16x32_bf16(false, Av, false, Bv, (short)0,
                                                              acc[nt], false, false);
        }
    }

    int nlo = lane & 15;
    int mbase = (lane & 16) ? 8 : 0;
#pragma unroll
    for (int nt = 0; nt < 4; ++nt) {
        int cout = nt * 16 + nlo;
        float s = sc[cout], t = bi[cout];
#pragma unroll
        for (int j = 0; j < 8; ++j) {
            int q = m0 + mbase + j;
            float v = acc[nt][j] * s + t;
            if (doRelu) v = fmaxf(v, 0.f);
            if (obf) {
                obf[(size_t)q * 64 + cout] = f2bf(v);
            } else {
                int qn = q >> 14, qr = q & (IMG - 1);
                of32[((size_t)(qn * 64 + cout) << 14) + qr] = v;
            }
        }
    }
}

// ---------------- K3: per-pixel channel reduction for down1/down2 --------
__global__ __launch_bounds__(256) void tap_reduce_k(const unsigned short* __restrict__ x1,
                                                    const float* __restrict__ d1w,
                                                    const float* __restrict__ d2w,
                                                    float* __restrict__ u1, float* __restrict__ t2) {
    __shared__ float w1[576], w2[576];
    for (int i = threadIdx.x; i < 576; i += 256) { w1[i] = d1w[i]; w2[i] = d2w[i]; }
    __syncthreads();
    int p = blockIdx.x * 256 + threadIdx.x;
    if (p >= P_TOT) return;
    float a1[9], a2[9];
#pragma unroll
    for (int t = 0; t < 9; ++t) { a1[t] = 0.f; a2[t] = 0.f; }
    const uint4* r4 = (const uint4*)(x1 + (size_t)p * 64);
#pragma unroll
    for (int cc = 0; cc < 8; ++cc) {
        uint4 q = r4[cc];
        unsigned qs[4] = {q.x, q.y, q.z, q.w};
#pragma unroll
        for (int e = 0; e < 4; ++e) {
            int c = cc * 8 + e * 2;
            float v0 = bf2f((unsigned short)qs[e]);
            float v1 = bf2f((unsigned short)(qs[e] >> 16));
#pragma unroll
            for (int t = 0; t < 9; ++t) {
                a1[t] = fmaf(v0, w1[c * 9 + t], a1[t]);
                a2[t] = fmaf(v0, w2[c * 9 + t], a2[t]);
                a1[t] = fmaf(v1, w1[(c + 1) * 9 + t], a1[t]);
                a2[t] = fmaf(v1, w2[(c + 1) * 9 + t], a2[t]);
            }
        }
    }
#pragma unroll
    for (int t = 0; t < 9; ++t) { u1[t * P_TOT + p] = a1[t]; t2[t * P_TOT + p] = a2[t]; }
}

// ---------------- K4: x_in + fused upsample/down2/Haar -------------------
__global__ __launch_bounds__(256) void fuse_wavelet_k(const float* __restrict__ u1,
                                                      const float* __restrict__ t2,
                                                      const float* __restrict__ d1b,
                                                      const float* __restrict__ d2b,
                                                      float* __restrict__ xin,
                                                      float* __restrict__ cAm,
                                                      float* __restrict__ es) {
    int p = blockIdx.x * 256 + threadIdx.x;
    if (p >= P_TOT) return;
    int n_ = p >> 14, rem = p & (IMG - 1);
    int y = rem >> 7, x = rem & 127;

    // x_in = conv(x1, down1_w) + b : 9-point stencil on u1 maps
    float s = d1b[0];
#pragma unroll
    for (int t = 0; t < 9; ++t) {
        int iy = y + t / 3 - 1, ix = x + t % 3 - 1;
        if (iy >= 0 && iy < 128 && ix >= 0 && ix < 128)
            s += u1[t * P_TOT + (n_ << 14) + (iy << 7) + ix];
    }
    xin[p] = s;

    float b2 = d2b[0];
    float hv[4];
#pragma unroll
    for (int d = 0; d < 4; ++d) {
        int Y = 2 * y + (d >> 1), X = 2 * x + (d & 1);
        float acc = b2;
#pragma unroll
        for (int t = 0; t < 9; ++t) {
            int Yp = Y + t / 3 - 1, Xp = X + t % 3 - 1;
            if (Yp < 0 || Yp > 255 || Xp < 0 || Xp > 255) continue;
            float cy = Yp * (127.0f / 255.0f);
            float cx = Xp * (127.0f / 255.0f);
            int ly = (int)cy, lx = (int)cx;
            float wy = cy - ly, wx = cx - lx;
            int hy = min(ly + 1, 127), hx = min(lx + 1, 127);
            const float* tm = t2 + t * P_TOT + (n_ << 14);
            float v00 = tm[(ly << 7) + lx], v01 = tm[(ly << 7) + hx];
            float v10 = tm[(hy << 7) + lx], v11 = tm[(hy << 7) + hx];
            acc += (1.f - wy) * ((1.f - wx) * v00 + wx * v01) + wy * ((1.f - wx) * v10 + wx * v11);
        }
        hv[d] = acc;
    }
    float a = hv[0], b = hv[1], c = hv[2], dd = hv[3];
    cAm[p] = (a + b + c + dd) * 0.5f;
    float LH = (-a - b + c + dd) * 0.5f;
    float HL = (-a + b - c + dd) * 0.5f;
    float HH = (a - b - c + dd) * 0.5f;
    es[p] = fabsf(LH) + fabsf(HL) + fabsf(HH);
}

// ---------------- K5: wt = 3x3(cA)+b ; gab = sigmoid(mean gabor) ---------
__global__ __launch_bounds__(256) void wt_gabor_k(const float* __restrict__ cAm,
                                                  const float* __restrict__ xin,
                                                  const float* __restrict__ wtw,
                                                  const float* __restrict__ wtb,
                                                  float* __restrict__ wt, float* __restrict__ gab) {
    __shared__ float gk[196];
    __shared__ float ww[10];
    int tid = threadIdx.x;
    if (tid < 196) {
        int th = tid / 49, j = tid % 49;
        float yv = (float)(j / 7 - 3), xv = (float)(j % 7 - 3);
        float theta = th * 0.78539816339744831f;        // pi/4
        float ct = cosf(theta), st = sinf(theta);
        float xt = xv * ct + yv * st;
        float yt = -xv * st + yv * ct;
        gk[tid] = expf(-0.5f * (xt * xt / 16.f + yt * yt / 64.f)) *
                  cosf(6.2831853071795865f * xt / 10.f);
    }
    if (tid < 9) ww[tid] = wtw[tid];
    if (tid == 9) ww[9] = wtb[0];
    __syncthreads();

    int p = blockIdx.x * 256 + tid;
    if (p >= P_TOT) return;
    int n_ = p >> 14, rem = p & (IMG - 1);
    int y = rem >> 7, x = rem & 127;
    const float* cA0 = cAm + (n_ << 14);
    const float* xi0 = xin + (n_ << 14);

    float s = ww[9];
#pragma unroll
    for (int t = 0; t < 9; ++t) {
        int iy = y + t / 3 - 1, ix = x + t % 3 - 1;
        if (iy >= 0 && iy < 128 && ix >= 0 && ix < 128) s += ww[t] * cA0[(iy << 7) + ix];
    }
    wt[p] = s;

    float g = 0.f;
    for (int th = 0; th < 4; ++th) {
#pragma unroll
        for (int j = 0; j < 49; ++j) {
            int iy = y + j / 7 - 3, ix = x + j % 7 - 3;
            if (iy >= 0 && iy < 128 && ix >= 0 && ix < 128)
                g += gk[th * 49 + j] * xi0[(iy << 7) + ix];
        }
    }
    gab[p] = sigmoidf(0.25f * g);
}

// ---------------- K6: up-conv(bn2,relu) + edge-conv -> combine input -----
__global__ __launch_bounds__(256) void upfuse_k(const float* __restrict__ wt,
                                                const float* __restrict__ gab,
                                                const float* __restrict__ es,
                                                const unsigned short* __restrict__ x1b,
                                                const float* __restrict__ upw,  // 64*2*9
                                                const float* __restrict__ ew,   // 64*9
                                                const float* __restrict__ eb,
                                                const float* __restrict__ sc2,
                                                const float* __restrict__ bi2,
                                                unsigned short* __restrict__ cmb) {
    __shared__ float s_up[1152], s_ew[576], s_eb[64], s_s2[64], s_b2[64];
    for (int i = threadIdx.x; i < 1152; i += 256) s_up[i] = upw[i];
    for (int i = threadIdx.x; i < 576; i += 256) s_ew[i] = ew[i];
    if (threadIdx.x < 64) {
        s_eb[threadIdx.x] = eb[threadIdx.x];
        s_s2[threadIdx.x] = sc2[threadIdx.x];
        s_b2[threadIdx.x] = bi2[threadIdx.x];
    }
    __syncthreads();

    int t = blockIdx.x * 256 + threadIdx.x;
    if (t >= P_TOT * 64) return;
    int cout = t & 63;
    int p = t >> 6;
    int n_ = p >> 14, rem = p & (IMG - 1);
    int y = rem >> 7, x = rem & 127;
    int base = n_ << 14;

    float f = 0.f, e = s_eb[cout];
#pragma unroll
    for (int tp = 0; tp < 9; ++tp) {
        int iy = y + tp / 3 - 1, ix = x + tp % 3 - 1;
        if (iy >= 0 && iy < 128 && ix >= 0 && ix < 128) {
            int nb = base + (iy << 7) + ix;
            f += wt[nb] * s_up[cout * 18 + tp] + gab[nb] * s_up[cout * 18 + 9 + tp];
            e += es[nb] * s_ew[cout * 9 + tp];
        }
    }
    f = fmaxf(f * s_s2[cout] + s_b2[cout], 0.f);
    float xv = bf2f(x1b[(size_t)p * 64 + cout]);
    cmb[(size_t)p * 128 + cout]      = f2bf(f);
    cmb[(size_t)p * 128 + 64 + cout] = f2bf(e * (xv + 1.f));
}

// ---------------- K8: per-(n,c) sum & max over spatial -------------------
__global__ __launch_bounds__(256) void chanstat_k(const unsigned short* __restrict__ fus,
                                                  float* __restrict__ csum, float* __restrict__ cmax) {
    int n_ = blockIdx.x;                 // 16 blocks
    int c = threadIdx.x & 63;
    int ch = threadIdx.x >> 6;           // 4 chunks
    float s = 0.f, m = -1e30f;
    for (int i = ch; i < IMG; i += 4) {
        float v = bf2f(fus[((size_t)(n_ * IMG + i)) * 64 + c]);
        s += v;
        m = fmaxf(m, v);
    }
    __shared__ float ss[256], sm[256];
    ss[threadIdx.x] = s; sm[threadIdx.x] = m;
    __syncthreads();
    if (ch == 0) {
        for (int k = 1; k < 4; ++k) { s += ss[c + k * 64]; m = fmaxf(m, sm[c + k * 64]); }
        csum[n_ * 64 + c] = s;
        cmax[n_ * 64 + c] = m;
    }
}

// ---------------- K9: channel-attention MLP ------------------------------
__global__ void ca_k(const float* __restrict__ csum, const float* __restrict__ cmax,
                     const float* __restrict__ fc1, const float* __restrict__ fc2,
                     float* __restrict__ ca) {
    int n_ = blockIdx.x;                 // 16 blocks x 64 threads
    int c = threadIdx.x;
    __shared__ float avg[64], mx[64], h1[4], h2[4];
    avg[c] = csum[n_ * 64 + c] * (1.f / IMG);
    mx[c]  = cmax[n_ * 64 + c];
    __syncthreads();
    if (c < 4) {
        float a = 0.f, b = 0.f;
        for (int k = 0; k < 64; ++k) { a += fc1[c * 64 + k] * avg[k]; b += fc1[c * 64 + k] * mx[k]; }
        h1[c] = fmaxf(a, 0.f); h2[c] = fmaxf(b, 0.f);
    }
    __syncthreads();
    float o = 0.f;
#pragma unroll
    for (int r = 0; r < 4; ++r) o += fc2[c * 4 + r] * (h1[r] + h2[r]);
    ca[n_ * 64 + c] = sigmoidf(o);
}

// ---------------- K10a: spatial mean/max of ca-scaled fusion -------------
__global__ __launch_bounds__(256) void spstat_k(const unsigned short* __restrict__ fus,
                                                const float* __restrict__ ca,
                                                float* __restrict__ spm, float* __restrict__ spx) {
    __shared__ float sca[64];
    int p0 = blockIdx.x * 256;
    int n_ = p0 >> 14;                    // uniform per block (16384 % 256 == 0)
    if (threadIdx.x < 64) sca[threadIdx.x] = ca[n_ * 64 + threadIdx.x];
    __syncthreads();
    int p = p0 + threadIdx.x;
    const uint4* r4 = (const uint4*)(fus + (size_t)p * 64);
    float s = 0.f, m = -1e30f;
#pragma unroll
    for (int cc = 0; cc < 8; ++cc) {
        uint4 q = r4[cc];
        unsigned qs[4] = {q.x, q.y, q.z, q.w};
#pragma unroll
        for (int e = 0; e < 4; ++e) {
            int c = cc * 8 + e * 2;
            float v0 = bf2f((unsigned short)qs[e]) * sca[c];
            float v1 = bf2f((unsigned short)(qs[e] >> 16)) * sca[c + 1];
            s += v0 + v1;
            m = fmaxf(m, fmaxf(v0, v1));
        }
    }
    spm[p] = s * (1.f / 64.f);
    spx[p] = m;
}

// ---------------- K10b: spatial attention + apply ------------------------
__global__ __launch_bounds__(256) void spatt_k(const unsigned short* __restrict__ fus,
                                               const float* __restrict__ ca,
                                               const float* __restrict__ spm,
                                               const float* __restrict__ spx,
                                               const float* __restrict__ spw,  // (1,2,7,7)
                                               unsigned short* __restrict__ att) {
    __shared__ float w[98];
    __shared__ float sca[64];
    int p0 = blockIdx.x * 256;
    int n_ = p0 >> 14;
    if (threadIdx.x < 98) w[threadIdx.x] = spw[threadIdx.x];
    if (threadIdx.x >= 128 && threadIdx.x < 192) sca[threadIdx.x - 128] = ca[n_ * 64 + threadIdx.x - 128];
    __syncthreads();
    int p = p0 + threadIdx.x;
    int rem = p & (IMG - 1);
    int y = rem >> 7, x = rem & 127;
    int base = n_ << 14;
    float g = 0.f;
#pragma unroll
    for (int j = 0; j < 49; ++j) {
        int iy = y + j / 7 - 3, ix = x + j % 7 - 3;
        if (iy >= 0 && iy < 128 && ix >= 0 && ix < 128) {
            int nb = base + (iy << 7) + ix;
            g += w[j] * spm[nb] + w[49 + j] * spx[nb];
        }
    }
    float sa = sigmoidf(g);
    const uint4* f4 = (const uint4*)(fus + (size_t)p * 64);
    uint4* o4 = (uint4*)(att + (size_t)p * 64);
#pragma unroll
    for (int cc = 0; cc < 8; ++cc) {
        uint4 q = f4[cc];
        unsigned qs[4] = {q.x, q.y, q.z, q.w};
        uint4 o;
        unsigned os[4];
#pragma unroll
        for (int e = 0; e < 4; ++e) {
            int c = cc * 8 + e * 2;
            float v0 = bf2f((unsigned short)qs[e]) * sca[c] * sa;
            float v1 = bf2f((unsigned short)(qs[e] >> 16)) * sca[c + 1] * sa;
            os[e] = (unsigned)f2bf(v0) | ((unsigned)f2bf(v1) << 16);
        }
        o.x = os[0]; o.y = os[1]; o.z = os[2]; o.w = os[3];
        o4[cc] = o;
    }
}

// =========================================================================
extern "C" void kernel_launch(void* const* d_in, const int* in_sizes, int n_in,
                              void* d_out, int out_size, void* d_ws, size_t ws_size,
                              hipStream_t stream) {
    const float* x        = (const float*)d_in[0];
    const float* conv_in_w= (const float*)d_in[1];
    const float* bn1_g = (const float*)d_in[2],  *bn1_b = (const float*)d_in[3];
    const float* bn1_m = (const float*)d_in[4],  *bn1_v = (const float*)d_in[5];
    const float* down1_w = (const float*)d_in[6],  *down1_b = (const float*)d_in[7];
    const float* down2_w = (const float*)d_in[8],  *down2_b = (const float*)d_in[9];
    const float* wt_w  = (const float*)d_in[10], *wt_b  = (const float*)d_in[11];
    const float* edge_w= (const float*)d_in[12], *edge_b= (const float*)d_in[13];
    const float* up_w  = (const float*)d_in[14];
    const float* bn2_g = (const float*)d_in[15], *bn2_b = (const float*)d_in[16];
    const float* bn2_m = (const float*)d_in[17], *bn2_v = (const float*)d_in[18];
    const float* combine_w = (const float*)d_in[19], *combine_b = (const float*)d_in[20];
    const float* fc1_w = (const float*)d_in[21], *fc2_w = (const float*)d_in[22];
    const float* sp_w  = (const float*)d_in[23];
    const float* out_w = (const float*)d_in[24];
    const float* bn3_g = (const float*)d_in[25], *bn3_b = (const float*)d_in[26];
    const float* bn3_m = (const float*)d_in[27], *bn3_v = (const float*)d_in[28];

    char* ws = (char*)d_ws;
    unsigned short* xb   = (unsigned short*)(ws + 0);            // 32MB; reused as fus
    unsigned short* x1b  = (unsigned short*)(ws + 32 * MBYTES);  // 32MB; reused as att
    unsigned short* cmb  = (unsigned short*)(ws + 64 * MBYTES);  // 64MB
    float* u1   = (float*)(ws + 128 * MBYTES);                   // 9MB; reused as spm/spx
    float* t2   = (float*)(ws + 137 * MBYTES);                   // 9MB
    float* xin  = (float*)(ws + 146 * MBYTES);
    float* cAm  = (float*)(ws + 147 * MBYTES);
    float* es   = (float*)(ws + 148 * MBYTES);
    float* wtm  = (float*)(ws + 149 * MBYTES);
    float* gabm = (float*)(ws + 150 * MBYTES);
    unsigned short* w_in  = (unsigned short*)(ws + 151 * MBYTES);               // 72KB
    unsigned short* w_out = (unsigned short*)(ws + 151 * MBYTES + 128 * 1024);  // 72KB
    unsigned short* w_cmb = (unsigned short*)(ws + 151 * MBYTES + 256 * 1024);  // 144KB
    float* prm  = (float*)(ws + 152 * MBYTES);                   // 8x64 params
    float* csum = prm + 512;
    float* cmaxv= csum + 1024;
    float* cav  = cmaxv + 1024;
    unsigned short* fus  = xb;    // alias: xb free after conv_in
    unsigned short* attb = x1b;   // alias: x1b free after upfuse
    float* spm = u1;              // alias: u1/t2 free after wavelet
    float* spx = u1 + P_TOT;

    const float* sc1 = prm + 0 * 64,  *bi1 = prm + 1 * 64;
    const float* sc2 = prm + 2 * 64,  *bi2 = prm + 3 * 64;
    const float* sc3 = prm + 4 * 64,  *bi3 = prm + 5 * 64;
    const float* scU = prm + 6 * 64,  *biU = prm + 7 * 64;

    params_k<<<1, 64, 0, stream>>>(bn1_g, bn1_b, bn1_m, bn1_v,
                                   bn2_g, bn2_b, bn2_m, bn2_v,
                                   bn3_g, bn3_b, bn3_m, bn3_v, combine_b, prm);
    pack_w_k<<<144, 256, 0, stream>>>(conv_in_w, w_in, 64);
    pack_w_k<<<144, 256, 0, stream>>>(out_w, w_out, 64);
    pack_w_k<<<288, 256, 0, stream>>>(combine_w, w_cmb, 128);
    nchw2nhwc_k<<<4096, 256, 0, stream>>>(x, xb);

    // conv_in + bn1 + relu  -> x1 (NHWC bf16)
    conv3x3_wmma<64><<<2048, 256, 0, stream>>>(xb, w_in, sc1, bi1, x1b, nullptr, 1);

    tap_reduce_k<<<1024, 256, 0, stream>>>(x1b, down1_w, down2_w, u1, t2);
    fuse_wavelet_k<<<1024, 256, 0, stream>>>(u1, t2, down1_b, down2_b, xin, cAm, es);
    wt_gabor_k<<<1024, 256, 0, stream>>>(cAm, xin, wt_w, wt_b, wtm, gabm);
    upfuse_k<<<65536, 256, 0, stream>>>(wtm, gabm, es, x1b, up_w, edge_w, edge_b, sc2, bi2, cmb);

    // combine conv (128 -> 64) + bias -> fusion (NHWC bf16)
    conv3x3_wmma<128><<<2048, 256, 0, stream>>>(cmb, w_cmb, scU, biU, fus, nullptr, 0);

    chanstat_k<<<16, 256, 0, stream>>>(fus, csum, cmaxv);
    ca_k<<<16, 64, 0, stream>>>(csum, cmaxv, fc1_w, fc2_w, cav);
    spstat_k<<<1024, 256, 0, stream>>>(fus, cav, spm, spx);
    spatt_k<<<1024, 256, 0, stream>>>(fus, cav, spm, spx, sp_w, attb);

    // out conv + bn3 + relu -> d_out (NCHW f32)
    conv3x3_wmma<64><<<2048, 256, 0, stream>>>(attb, w_out, sc3, bi3, nullptr, (float*)d_out, 1);
}
